// FeedForwardMoE_25460566131186
// MI455X (gfx1250) — matrix operand order repaired and checked by reading, not compiled, not run on hardware
//
#include <hip/hip_runtime.h>
#include <hip/hip_bf16.h>
#include <math.h>
#include <stdint.h>

#define T_TOK 4096
#define D_DIM 1024
#define E_EXP 8
#define K_TOP 2
#define DFF   4096
#define STRIDE 40   // 80B rows: 16B-aligned b128 frags, conflict-free reads (20 dw, 16 rows distinct banks)

typedef __attribute__((ext_vector_type(16))) __bf16 v16bf;
typedef __attribute__((ext_vector_type(8)))  float  v8f;

union Frag { v16bf v; unsigned int u[8]; uint4 q[2]; };
union Acc  { v8f   v; float        f[8]; };

__device__ __forceinline__ unsigned short f2bf(float f) {
    unsigned int u = __float_as_uint(f);
    unsigned int r = u + 0x7FFFu + ((u >> 16) & 1u);   // RNE
    return (unsigned short)(r >> 16);
}

__device__ __forceinline__ unsigned pack2bf(float lo, float hi) {
    return (unsigned)f2bf(lo) | ((unsigned)f2bf(hi) << 16);
}

__device__ __forceinline__ float gelu_exact(float v) {
    return 0.5f * v * (1.0f + erff(v * 0.70710678118654752f));
}

// CDNA5 async copy: global -> LDS, 16B per lane, tracked by ASYNCcnt.
__device__ __forceinline__ void async_copy_b128(unsigned lds_off, const void* gptr) {
    asm volatile("global_load_async_to_lds_b128 %0, %1, off"
                 :: "v"(lds_off), "v"(gptr) : "memory");
}
__device__ __forceinline__ void wait_async0() {
    asm volatile("s_wait_asynccnt 0" ::: "memory");
}

// ---------------------------------------------------------------------------
// init: convert x -> bf16, zero expert counters
// ---------------------------------------------------------------------------
__global__ __launch_bounds__(256)
void moe_init_kernel(const float* __restrict__ x,
                     unsigned short* __restrict__ xb,
                     int* __restrict__ counts) {
    int base = blockIdx.x * 1024 + threadIdx.x * 4;
    float4 v = *(const float4*)(x + base);
    union { unsigned short s[4]; uint2 u; } pk;
    pk.s[0] = f2bf(v.x); pk.s[1] = f2bf(v.y); pk.s[2] = f2bf(v.z); pk.s[3] = f2bf(v.w);
    *(uint2*)(xb + base) = pk.u;
    if (blockIdx.x == 0 && threadIdx.x < E_EXP) counts[threadIdx.x] = 0;
}

// ---------------------------------------------------------------------------
// gating: one wave per token; softmax, top-2, renormalize, append to lists
// ---------------------------------------------------------------------------
__global__ __launch_bounds__(256)
void moe_gate_kernel(const float* __restrict__ x,
                     const float* __restrict__ Wg,
                     const float* __restrict__ bg,
                     int* __restrict__ counts,
                     int* __restrict__ perm,
                     float* __restrict__ pw,
                     int* __restrict__ sel) {
    int wave = threadIdx.x >> 5;
    int lane = threadIdx.x & 31;
    int t = blockIdx.x * 8 + wave;

    float acc[E_EXP];
#pragma unroll
    for (int e = 0; e < E_EXP; ++e) acc[e] = 0.f;

    const float* xt = x + (size_t)t * D_DIM;
    for (int d = lane; d < D_DIM; d += 32) {
        float xv = xt[d];
        const float4* wg = (const float4*)(Wg + (size_t)d * E_EXP);
        float4 g0 = wg[0], g1 = wg[1];
        acc[0] += xv * g0.x; acc[1] += xv * g0.y;
        acc[2] += xv * g0.z; acc[3] += xv * g0.w;
        acc[4] += xv * g1.x; acc[5] += xv * g1.y;
        acc[6] += xv * g1.z; acc[7] += xv * g1.w;
    }
#pragma unroll
    for (int off = 16; off > 0; off >>= 1)
#pragma unroll
        for (int e = 0; e < E_EXP; ++e)
            acc[e] += __shfl_xor(acc[e], off, 32);

    if (lane == 0) {
        float l[E_EXP], mx = -1e30f;
#pragma unroll
        for (int e = 0; e < E_EXP; ++e) { l[e] = acc[e] + bg[e]; mx = fmaxf(mx, l[e]); }
        float s = 0.f;
#pragma unroll
        for (int e = 0; e < E_EXP; ++e) { l[e] = expf(l[e] - mx); s += l[e]; }
        float inv = 1.f / s;
#pragma unroll
        for (int e = 0; e < E_EXP; ++e) l[e] *= inv;

        int i1 = 0; float m1 = l[0];
#pragma unroll
        for (int e = 1; e < E_EXP; ++e) if (l[e] > m1) { m1 = l[e]; i1 = e; }
        int i2 = (i1 == 0) ? 1 : 0; float m2 = l[i2];
#pragma unroll
        for (int e = 0; e < E_EXP; ++e)
            if (e != i1 && l[e] > m2) { m2 = l[e]; i2 = e; }

        float denom = m1 + m2 + 1e-8f;
        int s1 = atomicAdd(&counts[i1], 1);
        perm[i1 * T_TOK + s1] = t;  pw[i1 * T_TOK + s1] = m1 / denom;
        int s2 = atomicAdd(&counts[i2], 1);
        perm[i2 * T_TOK + s2] = t;  pw[i2 * T_TOK + s2] = m2 / denom;
        sel[t * 2 + 0] = (i1 << 16) | s1;
        sel[t * 2 + 1] = (i2 << 16) | s2;
    }
}

__global__ void moe_scan_kernel(const int* __restrict__ counts, int* __restrict__ basep) {
    if (threadIdx.x == 0) {
        int s = 0;
        for (int e = 0; e < E_EXP; ++e) { basep[e] = s; s += counts[e]; }
    }
}

// ---------------------------------------------------------------------------
// fragment helpers: two aligned b128 reads per fragment
// A row-major [m][k]; B column-major [n][k]; both stride 40 halves.
// ---------------------------------------------------------------------------
__device__ __forceinline__ Frag load_a_frag(const unsigned short (*L)[STRIDE], int m, int half) {
    Frag f;
    const char* row = (const char*)&L[m][0];
    f.q[0] = *(const uint4*)(row + 16 * half);        // K 8h..8h+7
    f.q[1] = *(const uint4*)(row + 32 + 16 * half);   // K 16+8h..23+8h
    return f;
}

__device__ __forceinline__ Frag load_b_frag(const unsigned short (*L)[STRIDE], int n, int half) {
    Frag f;
    const char* col = (const char*)&L[n][0];
    f.q[0] = *(const uint4*)(col + 16 * half);        // K 8h..8h+7
    f.q[1] = *(const uint4*)(col + 32 + 16 * half);   // K 16+8h..23+8h
    return f;
}

#define WMMA_BF16(ACC, A, B)                                                   \
    (ACC).v = __builtin_amdgcn_wmma_f32_16x16x32_bf16(                         \
        false, (A).v, false, (B).v, (short)0, (ACC).v, false, false)

// ---------------------------------------------------------------------------
// GEMM macro body: block tile 128(M) x 128(N), K-step 32, double-buffered.
// 8 waves as 4(M) x 2(N); wave tile 32x64 -> 2 A frags, 4 B frags, 8 WMMAs.
// A staged async (bf16); B staged fp32 -> packed bf16 pairs.
// ---------------------------------------------------------------------------
__global__ __launch_bounds__(256)
void moe_gemm1_kernel(const unsigned short* __restrict__ xb,
                      const float* __restrict__ W1,
                      const float* __restrict__ bias1,
                      const int* __restrict__ counts,
                      const int* __restrict__ basep,
                      const int* __restrict__ perm,
                      unsigned short* __restrict__ H) {
    int e  = blockIdx.z;
    int ne = counts[e];
    int m0 = blockIdx.y * 128;
    if (m0 >= ne) return;
    int n0 = blockIdx.x * 128;
    int hbase = basep[e];

    __shared__ unsigned short As[2][128][STRIDE];
    __shared__ unsigned short Bs[2][128][STRIDE];   // column-major [n][k]

    int tid  = threadIdx.x;
    int lane = tid & 31;
    int wave = tid >> 5;
    int wm = wave & 3, wn = wave >> 2;
    int half = lane >> 4, ml = lane & 15;

    // A staging: row = tid/2, 16 bf16 (2 async b128) per thread
    int arow = tid >> 1;
    int aseg = (tid & 1) * 16;
    int aslot = m0 + arow; if (aslot > ne - 1) aslot = ne - 1;
    const unsigned short* asrc =
        xb + (size_t)perm[e * T_TOK + aslot] * D_DIM + aseg;

    // B staging: 2 k-rows x 8 n per thread
    int kp = tid >> 4;          // k pair 0..15 -> rows 2kp, 2kp+1
    int cg = (tid & 15) * 8;    // col group
    const float* bsrc = W1 + (size_t)e * D_DIM * DFF + (size_t)(2 * kp) * DFF + n0 + cg;

    Acc acc[2][4];
#pragma unroll
    for (int a = 0; a < 2; ++a)
#pragma unroll
        for (int i = 0; i < 4; ++i)
#pragma unroll
            for (int j = 0; j < 8; ++j) acc[a][i].f[j] = 0.f;

    unsigned lds_a[2];
    lds_a[0] = (unsigned)(uintptr_t)&As[0][arow][aseg];
    lds_a[1] = (unsigned)(uintptr_t)&As[1][arow][aseg];

    float4 r0a, r0b, r1a, r1b;
    // prologue: stage K-step 0 into buffer 0
    async_copy_b128(lds_a[0],      asrc);
    async_copy_b128(lds_a[0] + 16, asrc + 8);
    r0a = *(const float4*)(bsrc);       r0b = *(const float4*)(bsrc + 4);
    r1a = *(const float4*)(bsrc + DFF); r1b = *(const float4*)(bsrc + DFF + 4);
    {
        unsigned* bd = (unsigned*)&Bs[0][cg][2 * kp];
        const int cs = STRIDE / 2;  // dwords per column
        bd[0 * cs] = pack2bf(r0a.x, r1a.x); bd[1 * cs] = pack2bf(r0a.y, r1a.y);
        bd[2 * cs] = pack2bf(r0a.z, r1a.z); bd[3 * cs] = pack2bf(r0a.w, r1a.w);
        bd[4 * cs] = pack2bf(r0b.x, r1b.x); bd[5 * cs] = pack2bf(r0b.y, r1b.y);
        bd[6 * cs] = pack2bf(r0b.z, r1b.z); bd[7 * cs] = pack2bf(r0b.w, r1b.w);
    }
    wait_async0();
    __syncthreads();

    const int NSTEP = D_DIM / 32;
    for (int s = 0; s < NSTEP; ++s) {
        int cur = s & 1, nxt = cur ^ 1;
        bool more = (s + 1 < NSTEP);
        if (more) {
            const unsigned short* ap = asrc + (s + 1) * 32;
            async_copy_b128(lds_a[nxt],      ap);
            async_copy_b128(lds_a[nxt] + 16, ap + 8);
            const float* bp = bsrc + (size_t)(s + 1) * 32 * DFF;
            r0a = *(const float4*)(bp);       r0b = *(const float4*)(bp + 4);
            r1a = *(const float4*)(bp + DFF); r1b = *(const float4*)(bp + DFF + 4);
        }
        Frag a0 = load_a_frag(As[cur], (wm << 5) + ml,      half);
        Frag a1 = load_a_frag(As[cur], (wm << 5) + 16 + ml, half);
        Frag b0 = load_b_frag(Bs[cur], (wn << 6) + ml,      half);
        Frag b1 = load_b_frag(Bs[cur], (wn << 6) + 16 + ml, half);
        Frag b2 = load_b_frag(Bs[cur], (wn << 6) + 32 + ml, half);
        Frag b3 = load_b_frag(Bs[cur], (wn << 6) + 48 + ml, half);
        WMMA_BF16(acc[0][0], a0, b0); WMMA_BF16(acc[1][0], a1, b0);
        WMMA_BF16(acc[0][1], a0, b1); WMMA_BF16(acc[1][1], a1, b1);
        WMMA_BF16(acc[0][2], a0, b2); WMMA_BF16(acc[1][2], a1, b2);
        WMMA_BF16(acc[0][3], a0, b3); WMMA_BF16(acc[1][3], a1, b3);
        if (more) {
            unsigned* bd = (unsigned*)&Bs[nxt][cg][2 * kp];
            const int cs = STRIDE / 2;
            bd[0 * cs] = pack2bf(r0a.x, r1a.x); bd[1 * cs] = pack2bf(r0a.y, r1a.y);
            bd[2 * cs] = pack2bf(r0a.z, r1a.z); bd[3 * cs] = pack2bf(r0a.w, r1a.w);
            bd[4 * cs] = pack2bf(r0b.x, r1b.x); bd[5 * cs] = pack2bf(r0b.y, r1b.y);
            bd[6 * cs] = pack2bf(r0b.z, r1b.z); bd[7 * cs] = pack2bf(r0b.w, r1b.w);
            wait_async0();
            __syncthreads();
        }
    }

    size_t bb = (size_t)e * DFF + n0 + (wn << 6);
#pragma unroll
    for (int sub = 0; sub < 2; ++sub)
#pragma unroll
        for (int j = 0; j < 8; ++j) {
            int r = m0 + (wm << 5) + (sub << 4) + (half << 3) + j;
            if (r < ne) {
                size_t hrow = (size_t)(hbase + r) * DFF + n0 + (wn << 6);
#pragma unroll
                for (int i = 0; i < 4; ++i) {
                    int n = (i << 4) + ml;
                    float v = gelu_exact(acc[sub][i].f[j] + bias1[bb + n]);
                    H[hrow + n] = f2bf(v);
                }
            }
        }
}

__global__ __launch_bounds__(256)
void moe_gemm2_kernel(const unsigned short* __restrict__ H,
                      const float* __restrict__ W2,
                      const float* __restrict__ bias2,
                      const int* __restrict__ counts,
                      const int* __restrict__ basep,
                      float* __restrict__ Yb) {
    int e  = blockIdx.z;
    int ne = counts[e];
    int m0 = blockIdx.y * 128;
    if (m0 >= ne) return;
    int n0 = blockIdx.x * 128;
    int hbase = basep[e];

    __shared__ unsigned short As[2][128][STRIDE];
    __shared__ unsigned short Bs[2][128][STRIDE];

    int tid  = threadIdx.x;
    int lane = tid & 31;
    int wave = tid >> 5;
    int wm = wave & 3, wn = wave >> 2;
    int half = lane >> 4, ml = lane & 15;

    int arow = tid >> 1;
    int aseg = (tid & 1) * 16;
    int aslot = m0 + arow; if (aslot > ne - 1) aslot = ne - 1;
    const unsigned short* asrc = H + (size_t)(hbase + aslot) * DFF + aseg;

    int kp = tid >> 4;
    int cg = (tid & 15) * 8;
    const float* bsrc = W2 + (size_t)e * DFF * D_DIM + (size_t)(2 * kp) * D_DIM + n0 + cg;

    Acc acc[2][4];
#pragma unroll
    for (int a = 0; a < 2; ++a)
#pragma unroll
        for (int i = 0; i < 4; ++i)
#pragma unroll
            for (int j = 0; j < 8; ++j) acc[a][i].f[j] = 0.f;

    unsigned lds_a[2];
    lds_a[0] = (unsigned)(uintptr_t)&As[0][arow][aseg];
    lds_a[1] = (unsigned)(uintptr_t)&As[1][arow][aseg];

    float4 r0a, r0b, r1a, r1b;
    async_copy_b128(lds_a[0],      asrc);
    async_copy_b128(lds_a[0] + 16, asrc + 8);
    r0a = *(const float4*)(bsrc);         r0b = *(const float4*)(bsrc + 4);
    r1a = *(const float4*)(bsrc + D_DIM); r1b = *(const float4*)(bsrc + D_DIM + 4);
    {
        unsigned* bd = (unsigned*)&Bs[0][cg][2 * kp];
        const int cs = STRIDE / 2;
        bd[0 * cs] = pack2bf(r0a.x, r1a.x); bd[1 * cs] = pack2bf(r0a.y, r1a.y);
        bd[2 * cs] = pack2bf(r0a.z, r1a.z); bd[3 * cs] = pack2bf(r0a.w, r1a.w);
        bd[4 * cs] = pack2bf(r0b.x, r1b.x); bd[5 * cs] = pack2bf(r0b.y, r1b.y);
        bd[6 * cs] = pack2bf(r0b.z, r1b.z); bd[7 * cs] = pack2bf(r0b.w, r1b.w);
    }
    wait_async0();
    __syncthreads();

    const int NSTEP = DFF / 32;
    for (int s = 0; s < NSTEP; ++s) {
        int cur = s & 1, nxt = cur ^ 1;
        bool more = (s + 1 < NSTEP);
        if (more) {
            const unsigned short* ap = asrc + (s + 1) * 32;
            async_copy_b128(lds_a[nxt],      ap);
            async_copy_b128(lds_a[nxt] + 16, ap + 8);
            const float* bp = bsrc + (size_t)(s + 1) * 32 * D_DIM;
            r0a = *(const float4*)(bp);         r0b = *(const float4*)(bp + 4);
            r1a = *(const float4*)(bp + D_DIM); r1b = *(const float4*)(bp + D_DIM + 4);
        }
        Frag a0 = load_a_frag(As[cur], (wm << 5) + ml,      half);
        Frag a1 = load_a_frag(As[cur], (wm << 5) + 16 + ml, half);
        Frag b0 = load_b_frag(Bs[cur], (wn << 6) + ml,      half);
        Frag b1 = load_b_frag(Bs[cur], (wn << 6) + 16 + ml, half);
        Frag b2 = load_b_frag(Bs[cur], (wn << 6) + 32 + ml, half);
        Frag b3 = load_b_frag(Bs[cur], (wn << 6) + 48 + ml, half);
        WMMA_BF16(acc[0][0], a0, b0); WMMA_BF16(acc[1][0], a1, b0);
        WMMA_BF16(acc[0][1], a0, b1); WMMA_BF16(acc[1][1], a1, b1);
        WMMA_BF16(acc[0][2], a0, b2); WMMA_BF16(acc[1][2], a1, b2);
        WMMA_BF16(acc[0][3], a0, b3); WMMA_BF16(acc[1][3], a1, b3);
        if (more) {
            unsigned* bd = (unsigned*)&Bs[nxt][cg][2 * kp];
            const int cs = STRIDE / 2;
            bd[0 * cs] = pack2bf(r0a.x, r1a.x); bd[1 * cs] = pack2bf(r0a.y, r1a.y);
            bd[2 * cs] = pack2bf(r0a.z, r1a.z); bd[3 * cs] = pack2bf(r0a.w, r1a.w);
            bd[4 * cs] = pack2bf(r0b.x, r1b.x); bd[5 * cs] = pack2bf(r0b.y, r1b.y);
            bd[6 * cs] = pack2bf(r0b.z, r1b.z); bd[7 * cs] = pack2bf(r0b.w, r1b.w);
            wait_async0();
            __syncthreads();
        }
    }

    size_t bb = (size_t)e * D_DIM + n0 + (wn << 6);
#pragma unroll
    for (int sub = 0; sub < 2; ++sub)
#pragma unroll
        for (int j = 0; j < 8; ++j) {
            int r = m0 + (wm << 5) + (sub << 4) + (half << 3) + j;
            if (r < ne) {
                size_t yrow = (size_t)(hbase + r) * D_DIM + n0 + (wn << 6);
#pragma unroll
                for (int i = 0; i < 4; ++i) {
                    int n = (i << 4) + ml;
                    Yb[yrow + n] = acc[sub][i].f[j] + bias2[bb + n];
                }
            }
        }
}

// ---------------------------------------------------------------------------
// combine (top-2 weighted sum of Yb rows) + LayerNorm, fused; block per token
// ---------------------------------------------------------------------------
__global__ __launch_bounds__(256)
void moe_combine_ln_kernel(const float* __restrict__ Yb,
                           const int* __restrict__ sel,
                           const float* __restrict__ pw,
                           const int* __restrict__ basep,
                           const float* __restrict__ gamma,
                           const float* __restrict__ beta,
                           float* __restrict__ out) {
    int t = blockIdx.x;
    int lane = threadIdx.x & 31;
    int wave = threadIdx.x >> 5;

    int a0 = sel[t * 2], a1 = sel[t * 2 + 1];
    int e0 = a0 >> 16, s0 = a0 & 0xFFFF;
    int e1 = a1 >> 16, s1 = a1 & 0xFFFF;
    int r0 = basep[e0] + s0;
    int r1 = basep[e1] + s1;
    float c0 = pw[e0 * T_TOK + s0];
    float c1 = pw[e1 * T_TOK + s1];

    int n = threadIdx.x * 4;
    float4 y0 = *(const float4*)(Yb + (size_t)r0 * D_DIM + n);
    float4 y1 = *(const float4*)(Yb + (size_t)r1 * D_DIM + n);
    float4 v;
    v.x = c0 * y0.x + c1 * y1.x;
    v.y = c0 * y0.y + c1 * y1.y;
    v.z = c0 * y0.z + c1 * y1.z;
    v.w = c0 * y0.w + c1 * y1.w;

    float s  = v.x + v.y + v.z + v.w;
    float sq = v.x * v.x + v.y * v.y + v.z * v.z + v.w * v.w;
#pragma unroll
    for (int off = 16; off > 0; off >>= 1) {
        s  += __shfl_xor(s,  off, 32);
        sq += __shfl_xor(sq, off, 32);
    }
    __shared__ float ss[8], ssq[8];
    if (lane == 0) { ss[wave] = s; ssq[wave] = sq; }
    __syncthreads();
    if (wave == 0) {
        float a = (lane < 8) ? ss[lane]  : 0.f;
        float b = (lane < 8) ? ssq[lane] : 0.f;
#pragma unroll
        for (int off = 4; off > 0; off >>= 1) {
            a += __shfl_xor(a, off, 32);
            b += __shfl_xor(b, off, 32);
        }
        if (lane == 0) {
            float mu  = a / (float)D_DIM;
            float var = b / (float)D_DIM - mu * mu;
            ss[0]  = mu;
            ssq[0] = rsqrtf(var + 1e-5f);
        }
    }
    __syncthreads();
    float mu = ss[0], rs = ssq[0];
    float4 g  = *(const float4*)(gamma + n);
    float4 be = *(const float4*)(beta + n);
    float4 o;
    o.x = (v.x - mu) * rs * g.x + be.x;
    o.y = (v.y - mu) * rs * g.y + be.y;
    o.z = (v.z - mu) * rs * g.z + be.z;
    o.w = (v.w - mu) * rs * g.w + be.w;
    *(float4*)(out + (size_t)t * D_DIM + n) = o;
}

// ---------------------------------------------------------------------------
extern "C" void kernel_launch(void* const* d_in, const int* in_sizes, int n_in,
                              void* d_out, int out_size, void* d_ws, size_t ws_size,
                              hipStream_t stream) {
    (void)in_sizes; (void)n_in; (void)out_size; (void)ws_size;
    const float* x     = (const float*)d_in[0];
    const float* Wg    = (const float*)d_in[1];
    const float* bg    = (const float*)d_in[2];
    const float* W1    = (const float*)d_in[3];
    const float* b1    = (const float*)d_in[4];
    const float* W2    = (const float*)d_in[5];
    const float* b2    = (const float*)d_in[6];
    const float* gamma = (const float*)d_in[7];
    const float* beta  = (const float*)d_in[8];
    float* out = (float*)d_out;

    char* ws = (char*)d_ws;
    size_t off = 0;
    auto alloc = [&](size_t bytes) -> void* {
        void* p = ws + off;
        off = (off + bytes + 255) & ~(size_t)255;
        return p;
    };
    unsigned short* xb   = (unsigned short*)alloc((size_t)T_TOK * D_DIM * 2);        // 8 MB
    unsigned short* H    = (unsigned short*)alloc((size_t)T_TOK * K_TOP * DFF * 2);  // 64 MB
    float*          Yb   = (float*)alloc((size_t)T_TOK * K_TOP * D_DIM * 4);         // 32 MB
    int*    counts = (int*)alloc(E_EXP * 4);
    int*    basep  = (int*)alloc(E_EXP * 4);
    int*    perm   = (int*)alloc((size_t)E_EXP * T_TOK * 4);
    float*  pw     = (float*)alloc((size_t)E_EXP * T_TOK * 4);
    int*    sel    = (int*)alloc((size_t)T_TOK * 2 * 4);

    moe_init_kernel<<<4096, 256, 0, stream>>>(x, xb, counts);
    moe_gate_kernel<<<T_TOK / 8, 256, 0, stream>>>(x, Wg, bg, counts, perm, pw, sel);
    moe_scan_kernel<<<1, 32, 0, stream>>>(counts, basep);
    moe_gemm1_kernel<<<dim3(DFF / 128, T_TOK / 128, E_EXP), 256, 0, stream>>>(
        xb, W1, b1, counts, basep, perm, H);
    moe_gemm2_kernel<<<dim3(D_DIM / 128, T_TOK / 128, E_EXP), 256, 0, stream>>>(
        H, W2, b2, counts, basep, Yb);
    moe_combine_ln_kernel<<<T_TOK, 256, 0, stream>>>(Yb, sel, pw, basep, gamma, beta, out);
}
